// WaveletPyramidCrossAttention_6966436954357
// MI455X (gfx1250) — matrix operand order repaired and checked by reading, not compiled, not run on hardware
//
#include <hip/hip_runtime.h>
#include <hip/hip_bf16.h>
#include <math.h>

// ---------------------------------------------------------------------------
// WaveletPyramidCrossAttention on gfx1250 (MI455X), WMMA f16 -> f32 path.
// B=2, N=4, C=256, H=W=64, heads=8, hd=32. Wave32, v_wmma_f32_16x16x32_f16.
// ---------------------------------------------------------------------------

typedef __attribute__((ext_vector_type(16))) _Float16 v16h;
typedef __attribute__((ext_vector_type(8)))  _Float16 v8h;
typedef __attribute__((ext_vector_type(8)))  float    v8f;

union Frag16 { v16h v; v8h h[2]; };
union F8     { v8f  v; float f[8]; };

#define C_DIM   256
#define HEADS   8
#define HD      32
#define POS     1024          // 32x32 after DWT
#define NKEY    4096          // 4 * 1024
#define ATT_SCALE 0.17677669529663687f  // 1/sqrt(32)

// ---------------------------------------------------------------------------
// Weight conversion f32 -> f16
// ---------------------------------------------------------------------------
__global__ void cvt_w_kernel(const float* __restrict__ W, _Float16* __restrict__ Wh) {
    int i = blockIdx.x * blockDim.x + threadIdx.x;
    if (i < C_DIM * C_DIM) Wh[i] = (_Float16)W[i];
}

__global__ void zero_stats_kernel(float* stats) {
    int i = threadIdx.x;
    if (i < 128) stats[i] = 0.0f;
}

// ---------------------------------------------------------------------------
// Haar DWT of query: -> ll (pos-major f16 [b][1024][256]) + yh f32 [b][c][3][1024]
// ---------------------------------------------------------------------------
__global__ void dwt_q_kernel(const float* __restrict__ q,
                             _Float16* __restrict__ llq,
                             float* __restrict__ yh) {
    int idx = blockIdx.x * blockDim.x + threadIdx.x;   // b*C*1024
    if (idx >= 2 * C_DIM * POS) return;
    int pos = idx & 1023;
    int c   = (idx >> 10) & 255;
    int b   = idx >> 18;
    int y = pos >> 5, x = pos & 31;
    const float* base = q + (((size_t)b * C_DIM + c) * 64 + 2 * y) * 64 + 2 * x;
    float a = base[0], bb = base[1], cc = base[64], dd = base[65];
    float ll = (a + bb + cc + dd) * 0.5f;
    float lh = (a - bb + cc - dd) * 0.5f;
    float hl = (a + bb - cc - dd) * 0.5f;
    float hh = (a - bb - cc + dd) * 0.5f;
    llq[((size_t)b * POS + pos) * C_DIM + c] = (_Float16)ll;
    size_t yb = (((size_t)b * C_DIM + c) * 3) * POS + pos;
    yh[yb]          = lh;
    yh[yb + POS]    = hl;
    yh[yb + 2*POS]  = hh;
}

// DWT of key_value_multi (8 images): only ll, pos-major f16 [img][1024][256]
__global__ void dwt_kv_kernel(const float* __restrict__ kv,
                              _Float16* __restrict__ llkv) {
    int idx = blockIdx.x * blockDim.x + threadIdx.x;   // img*C*1024
    if (idx >= 8 * C_DIM * POS) return;
    int pos = idx & 1023;
    int c   = (idx >> 10) & 255;
    int img = idx >> 18;
    int y = pos >> 5, x = pos & 31;
    const float* base = kv + (((size_t)img * C_DIM + c) * 64 + 2 * y) * 64 + 2 * x;
    float a = base[0], bb = base[1], cc = base[64], dd = base[65];
    llkv[((size_t)img * POS + pos) * C_DIM + c] = (_Float16)((a + bb + cc + dd) * 0.5f);
}

// ---------------------------------------------------------------------------
// conv1x1 GEMM via WMMA: out[m=pos][n=outch] = sum_k Xt[pos][k] * W[n][k] + b[n]
// A-frag: lanes 0-15 row M=lane, K={0..7,16..23}; lanes 16-31 row M=lane-16,
//         K={8..15,24..31}  -> two b128 loads at (kc+aoff) and (kc+16+aoff)
// B-frag: lane L: N=L&15, K same as A-frag  -> two b128 loads at (kc+boff) and (kc+16+boff)
// Two independent accumulators break the WMMA RAW chain on C.
// MODE 0: Q  (scaled, [bh][p][d] f16)     MODE 1: K ([bh][key][d] f16)
// MODE 2: V  transposed ([bh][d][key])    MODE 3: Wp proj (f32 [b][p][c])
// ---------------------------------------------------------------------------
template <int MODE>
__global__ void gemm_conv1x1_kernel(const _Float16* __restrict__ Xt,
                                    const _Float16* __restrict__ Wh,
                                    const float* __restrict__ bias,
                                    _Float16* __restrict__ outH,
                                    float* __restrict__ outF) {
    int lane = threadIdx.x & 31;
    int wave = threadIdx.x >> 5;
    int tile = blockIdx.x * (blockDim.x >> 5) + wave; // 1024 tiles per image
    int img   = tile >> 10;
    int t     = tile & 1023;
    int mbase = (t >> 4) * 16;
    int obase = (t & 15) * 16;
    int half  = lane >> 4;
    int lr    = lane & 15;

    const _Float16* xrow = Xt + ((size_t)img * POS + mbase + lr) * C_DIM;
    const _Float16* wrow = Wh + (size_t)(obase + lr) * C_DIM;
    int aoff = half ? 8 : 0;
    int boff = half ? 8 : 0;

    F8 acc0, acc1;
    acc0.v = (v8f)(0.0f);
    acc1.v = (v8f)(0.0f);
#pragma unroll
    for (int kc = 0; kc < C_DIM; kc += 64) {
        Frag16 a0, b0, a1, b1;
        a0.h[0] = *(const v8h*)(xrow + kc + aoff);
        a0.h[1] = *(const v8h*)(xrow + kc + 16 + aoff);
        b0.h[0] = *(const v8h*)(wrow + kc + boff);
        b0.h[1] = *(const v8h*)(wrow + kc + boff + 16);
        a1.h[0] = *(const v8h*)(xrow + kc + 32 + aoff);
        a1.h[1] = *(const v8h*)(xrow + kc + 48 + aoff);
        b1.h[0] = *(const v8h*)(wrow + kc + 32 + boff);
        b1.h[1] = *(const v8h*)(wrow + kc + 32 + boff + 16);
        acc0.v = __builtin_amdgcn_wmma_f32_16x16x32_f16(
            false, a0.v, false, b0.v, (short)0, acc0.v, false, false);
        acc1.v = __builtin_amdgcn_wmma_f32_16x16x32_f16(
            false, a1.v, false, b1.v, (short)0, acc1.v, false, false);
    }

    int ocol = obase + lr;            // output channel for this lane's column
    float bv = bias[ocol];
    int head = ocol >> 5;
    int d    = ocol & 31;

#pragma unroll
    for (int r = 0; r < 8; ++r) {
        int m = mbase + half * 8 + r;
        float val = acc0.f[r] + acc1.f[r] + bv;
        if (MODE == 0) {            // Q: [b*8+head][p][d], pre-scaled
            outH[(((size_t)img * HEADS + head) * POS + m) * HD + d] =
                (_Float16)(val * ATT_SCALE);
        } else if (MODE == 1) {     // K: [b*8+head][key][d]
            int b = img >> 2, nn = img & 3;
            int kidx = nn * POS + m;
            outH[(((size_t)b * HEADS + head) * NKEY + kidx) * HD + d] = (_Float16)val;
        } else if (MODE == 2) {     // V^T: [b*8+head][d][key]
            int b = img >> 2, nn = img & 3;
            int kidx = nn * POS + m;
            outH[(((size_t)b * HEADS + head) * HD + d) * NKEY + kidx] = (_Float16)val;
        } else {                    // Wp projection: f32 [b][p][c]
            outF[((size_t)img * POS + m) * C_DIM + ocol] = val;
        }
    }
}

// ---------------------------------------------------------------------------
// Flash attention: 4 waves/block share one (b,head); each wave owns a 16-row
// Q tile. K/V chunks of 64 keys are staged cooperatively in LDS (all waves in
// the block need identical K/V), then each wave runs 4 S-WMMAs, one online
// softmax round covering all 64 keys, and 4 accumulating P*V WMMAs.
// ---------------------------------------------------------------------------
#define KCHUNK 64
#define SKPAD  40   // 80 B row: 16B aligned, 20-dword bank stride (conflict-free /16)
#define SVPAD  72   // 144 B row: 16B aligned, 36-dword bank stride

__global__ void attn_flash_kernel(const _Float16* __restrict__ Qb,
                                  const _Float16* __restrict__ Kb,
                                  const _Float16* __restrict__ Vt,
                                  _Float16* __restrict__ Ob) {
    int tid  = threadIdx.x;
    int lane = tid & 31;
    int wave = tid >> 5;
    int bh    = blockIdx.x;                 // 0..15 (b*8+head)
    int ptile = blockIdx.y * 4 + wave;      // 0..63
    int b    = bh >> 3;
    int head = bh & 7;
    int half = lane >> 4;
    int lr   = lane & 15;
    int aoff = half ? 16 : 0;

    __shared__ _Float16 sK[KCHUNK][SKPAD];   // [key][d]      (padded rows)
    __shared__ _Float16 sV[HD][SVPAD];       // [d][key]      (padded rows)
    __shared__ _Float16 sP[4][16][KCHUNK];   // per-wave P staging

    // Q A-fragment (rows contiguous in d), resident all loop long
    const _Float16* qrow = Qb + (((size_t)bh) * POS + ptile * 16 + lr) * HD;
    Frag16 qa;
    qa.h[0] = *(const v8h*)(qrow + aoff);
    qa.h[1] = *(const v8h*)(qrow + aoff + 8);

    float mrun[8], lrun[8];
#pragma unroll
    for (int r = 0; r < 8; ++r) { mrun[r] = -1e30f; lrun[r] = 0.0f; }
    F8 o0, o1; o0.v = (v8f)(0.0f); o1.v = (v8f)(0.0f);
    v8f zero = (v8f)(0.0f);

    const _Float16* Kbase = Kb + (size_t)bh * NKEY * HD;
    const _Float16* Vbase = Vt + (size_t)bh * HD * NKEY;

    // cooperative-staging index split (128 threads, 16 halves each per buffer)
    int kk = tid >> 1, dh = (tid & 1) * 16;      // K: key row, d-half
    int dv = tid >> 2, ks = (tid & 3) * 16;      // V: d row, key quarter

    for (int kc = 0; kc < NKEY; kc += KCHUNK) {
        __syncthreads();   // previous chunk's LDS reads complete
        // stage K chunk [64 keys][32 d] and V chunk [32 d][64 keys]
        const _Float16* kg = Kbase + (size_t)(kc + kk) * HD + dh;
        *(v8h*)&sK[kk][dh]     = *(const v8h*)kg;
        *(v8h*)&sK[kk][dh + 8] = *(const v8h*)(kg + 8);
        const _Float16* vg = Vbase + (size_t)dv * NKEY + kc + ks;
        *(v8h*)&sV[dv][ks]     = *(const v8h*)vg;
        *(v8h*)&sV[dv][ks + 8] = *(const v8h*)(vg + 8);
        __syncthreads();

        // ---- S = Q K^T for 4 key tiles (independent WMMAs) ----
        F8 s[4];
#pragma unroll
        for (int j = 0; j < 4; ++j) {
            Frag16 kf;   // B[d][n]: key = j*16+lr, d = half*16 + h
            kf.h[0] = *(const v8h*)&sK[j * 16 + lr][half * 16];
            kf.h[1] = *(const v8h*)&sK[j * 16 + lr][half * 16 + 8];
            s[j].v = __builtin_amdgcn_wmma_f32_16x16x32_f16(
                false, qa.v, false, kf.v, (short)0, zero, false, false);
        }

        // ---- online softmax over all 64 keys in one reduction round ----
#pragma unroll
        for (int r = 0; r < 8; ++r) {
            float x0 = s[0].f[r], x1 = s[1].f[r], x2 = s[2].f[r], x3 = s[3].f[r];
            float tm = fmaxf(fmaxf(x0, x1), fmaxf(x2, x3));
#pragma unroll
            for (int off = 8; off; off >>= 1) tm = fmaxf(tm, __shfl_xor(tm, off));
            float nm = fmaxf(mrun[r], tm);
            float alpha = __expf(mrun[r] - nm);
            float p0 = __expf(x0 - nm);
            float p1 = __expf(x1 - nm);
            float p2 = __expf(x2 - nm);
            float p3 = __expf(x3 - nm);
            float rs = (p0 + p1) + (p2 + p3);
#pragma unroll
            for (int off = 8; off; off >>= 1) rs += __shfl_xor(rs, off);
            lrun[r] = lrun[r] * alpha + rs;
            mrun[r] = nm;
            o0.f[r] *= alpha;
            o1.f[r] *= alpha;
            int row = half * 8 + r;
            sP[wave][row][lr]      = (_Float16)p0;
            sP[wave][row][16 + lr] = (_Float16)p1;
            sP[wave][row][32 + lr] = (_Float16)p2;
            sP[wave][row][48 + lr] = (_Float16)p3;
        }

        // ---- O += P V : two 32-key A-frags x two 16-wide d tiles ----
        Frag16 pa0, pa1;
        pa0.h[0] = *(const v8h*)&sP[wave][lr][aoff];
        pa0.h[1] = *(const v8h*)&sP[wave][lr][aoff + 8];
        pa1.h[0] = *(const v8h*)&sP[wave][lr][32 + aoff];
        pa1.h[1] = *(const v8h*)&sP[wave][lr][32 + aoff + 8];

        Frag16 v00, v01, v10, v11;   // v[s][t]: key sub-chunk s, d tile t
        v00.h[0] = *(const v8h*)&sV[lr][half * 16];
        v00.h[1] = *(const v8h*)&sV[lr][half * 16 + 8];
        v01.h[0] = *(const v8h*)&sV[16 + lr][half * 16];
        v01.h[1] = *(const v8h*)&sV[16 + lr][half * 16 + 8];
        v10.h[0] = *(const v8h*)&sV[lr][32 + half * 16];
        v10.h[1] = *(const v8h*)&sV[lr][32 + half * 16 + 8];
        v11.h[0] = *(const v8h*)&sV[16 + lr][32 + half * 16];
        v11.h[1] = *(const v8h*)&sV[16 + lr][32 + half * 16 + 8];

        o0.v = __builtin_amdgcn_wmma_f32_16x16x32_f16(
            false, pa0.v, false, v00.v, (short)0, o0.v, false, false);
        o1.v = __builtin_amdgcn_wmma_f32_16x16x32_f16(
            false, pa0.v, false, v01.v, (short)0, o1.v, false, false);
        o0.v = __builtin_amdgcn_wmma_f32_16x16x32_f16(
            false, pa1.v, false, v10.v, (short)0, o0.v, false, false);
        o1.v = __builtin_amdgcn_wmma_f32_16x16x32_f16(
            false, pa1.v, false, v11.v, (short)0, o1.v, false, false);
    }

    // normalize and store O as [b][p][c] f16 (pos-major for the Wp GEMM)
#pragma unroll
    for (int r = 0; r < 8; ++r) {
        float inv = 1.0f / lrun[r];
        int row = ptile * 16 + half * 8 + r;
        int c0  = head * HD + lr;          // d-tile 0 column
        size_t base = ((size_t)b * POS + row) * C_DIM + c0;
        Ob[base]      = (_Float16)(o0.f[r] * inv);
        Ob[base + 16] = (_Float16)(o1.f[r] * inv);
    }
}

// ---------------------------------------------------------------------------
// IDWT + GroupNorm statistics (sum, sumsq per (b, group)) via block reduce
// ---------------------------------------------------------------------------
__global__ void idwt_gn_kernel(const float* __restrict__ Pout,
                               const float* __restrict__ yh,
                               float* __restrict__ ybuf,
                               float* __restrict__ stats) {
    int bc = blockIdx.x >> 2;
    int qq = blockIdx.x & 3;
    int b = bc >> 8, c = bc & 255;
    int pos = qq * 256 + threadIdx.x;
    int y = pos >> 5, x = pos & 31;

    float ll = Pout[((size_t)b * POS + pos) * C_DIM + c];
    size_t yb = (((size_t)b * C_DIM + c) * 3) * POS + pos;
    float lh = yh[yb], hl = yh[yb + POS], hh = yh[yb + 2 * POS];
    float a   = (ll + lh + hl + hh) * 0.5f;
    float bb2 = (ll - lh + hl - hh) * 0.5f;
    float cc  = (ll + lh - hl - hh) * 0.5f;
    float dd  = (ll - lh - hl + hh) * 0.5f;

    float* yo = ybuf + (((size_t)b * C_DIM + c) * 64 + 2 * y) * 64 + 2 * x;
    yo[0] = a; yo[1] = bb2; yo[64] = cc; yo[65] = dd;

    float s  = a + bb2 + cc + dd;
    float ss = a * a + bb2 * bb2 + cc * cc + dd * dd;

    __shared__ float rs[256], rss[256];
    rs[threadIdx.x] = s;  rss[threadIdx.x] = ss;
    __syncthreads();
    for (int st = 128; st; st >>= 1) {
        if (threadIdx.x < st) {
            rs[threadIdx.x]  += rs[threadIdx.x + st];
            rss[threadIdx.x] += rss[threadIdx.x + st];
        }
        __syncthreads();
    }
    if (threadIdx.x == 0) {
        int g = c >> 3;
        atomicAdd(&stats[(b * 32 + g) * 2],     rs[0]);
        atomicAdd(&stats[(b * 32 + g) * 2 + 1], rss[0]);
    }
}

// GroupNorm normalize + residual
__global__ void gn_final_kernel(const float* __restrict__ ybuf,
                                const float* __restrict__ stats,
                                const float* __restrict__ gamma,
                                const float* __restrict__ beta,
                                const float* __restrict__ query,
                                float* __restrict__ out) {
    size_t i = blockIdx.x * (size_t)blockDim.x + threadIdx.x;
    if (i >= (size_t)2 * C_DIM * 4096) return;
    int c = (int)((i >> 12) & 255);
    int b = (int)(i >> 20);
    int g = c >> 3;
    const float cnt = 8.0f * 4096.0f;
    float mean = stats[(b * 32 + g) * 2] / cnt;
    float var  = stats[(b * 32 + g) * 2 + 1] / cnt - mean * mean;
    float xn = (ybuf[i] - mean) * rsqrtf(var + 1e-5f);
    out[i] = xn * gamma[c] + beta[c] + query[i];
}

// ---------------------------------------------------------------------------
// Launch
// ---------------------------------------------------------------------------
extern "C" void kernel_launch(void* const* d_in, const int* in_sizes, int n_in,
                              void* d_out, int out_size, void* d_ws, size_t ws_size,
                              hipStream_t stream) {
    const float* query = (const float*)d_in[0];
    const float* kvm   = (const float*)d_in[1];
    // d_in[2] (value_multi) is unused by the reference
    const float* Wq = (const float*)d_in[3];  const float* bq = (const float*)d_in[4];
    const float* Wk = (const float*)d_in[5];  const float* bk = (const float*)d_in[6];
    const float* Wv = (const float*)d_in[7];  const float* bv = (const float*)d_in[8];
    const float* Wp = (const float*)d_in[9];  const float* bp = (const float*)d_in[10];
    const float* gamma = (const float*)d_in[11];
    const float* beta  = (const float*)d_in[12];
    float* out = (float*)d_out;

    // workspace carve-up (all 256B aligned), ~33 MB total
    char* ws = (char*)d_ws;
    size_t off = 0;
    auto alloc = [&](size_t bytes) -> void* {
        void* p = ws + off;
        off += (bytes + 255) & ~(size_t)255;
        return p;
    };
    _Float16* WqH   = (_Float16*)alloc(C_DIM * C_DIM * 2);
    _Float16* WkH   = (_Float16*)alloc(C_DIM * C_DIM * 2);
    _Float16* WvH   = (_Float16*)alloc(C_DIM * C_DIM * 2);
    _Float16* WpH   = (_Float16*)alloc(C_DIM * C_DIM * 2);
    _Float16* qllT  = (_Float16*)alloc((size_t)2 * POS * C_DIM * 2);
    _Float16* kvllT = (_Float16*)alloc((size_t)8 * POS * C_DIM * 2);
    float*    qyh   = (float*)   alloc((size_t)2 * C_DIM * 3 * POS * 4);
    _Float16* Qbuf  = (_Float16*)alloc((size_t)16 * POS * HD * 2);
    _Float16* Kbuf  = (_Float16*)alloc((size_t)16 * NKEY * HD * 2);
    _Float16* VTbuf = (_Float16*)alloc((size_t)16 * HD * NKEY * 2);
    _Float16* Obuf  = (_Float16*)alloc((size_t)2 * POS * C_DIM * 2);
    float*    Pout  = (float*)   alloc((size_t)2 * POS * C_DIM * 4);
    float*    ybuf  = (float*)   alloc((size_t)2 * C_DIM * 4096 * 4);
    float*    stats = (float*)   alloc(128 * 4);

    // 1. weight conversion + stat zeroing
    cvt_w_kernel<<<256, 256, 0, stream>>>(Wq, WqH);
    cvt_w_kernel<<<256, 256, 0, stream>>>(Wk, WkH);
    cvt_w_kernel<<<256, 256, 0, stream>>>(Wv, WvH);
    cvt_w_kernel<<<256, 256, 0, stream>>>(Wp, WpH);
    zero_stats_kernel<<<1, 128, 0, stream>>>(stats);

    // 2. DWT
    dwt_q_kernel<<<2048, 256, 0, stream>>>(query, qllT, qyh);
    dwt_kv_kernel<<<8192, 256, 0, stream>>>(kvm, kvllT);

    // 3. conv1x1 GEMMs (WMMA). 1024 tiles per image, 8 waves (tiles) per block.
    gemm_conv1x1_kernel<0><<<256,  256, 0, stream>>>(qllT,  WqH, bq, Qbuf,  nullptr);
    gemm_conv1x1_kernel<1><<<1024, 256, 0, stream>>>(kvllT, WkH, bk, Kbuf,  nullptr);
    gemm_conv1x1_kernel<2><<<1024, 256, 0, stream>>>(kvllT, WvH, bv, VTbuf, nullptr);

    // 4. flash attention (WMMA): 16 (b,h) x 64 q-tiles, 4 waves/block
    dim3 agrid(16, 16);
    attn_flash_kernel<<<agrid, 128, 0, stream>>>(Qbuf, Kbuf, VTbuf, Obuf);

    // 5. output projection (WMMA)
    gemm_conv1x1_kernel<3><<<256, 256, 0, stream>>>(Obuf, WpH, bp, nullptr, Pout);

    // 6. IDWT + GroupNorm stats, then normalize + residual
    idwt_gn_kernel<<<2048, 256, 0, stream>>>(Pout, qyh, ybuf, stats);
    gn_final_kernel<<<8192, 256, 0, stream>>>(ybuf, stats, gamma, beta, query, out);
}
